// WanI2VCrossAttention_20023137534388
// MI455X (gfx1250) — matrix operand order repaired and checked by reading, not compiled, not run on hardware
//
#include <hip/hip_runtime.h>
#include <math.h>

// ---------------------------------------------------------------------------
// WanI2V cross-attention for gfx1250 (MI455X).
//  - Projection / output GEMMs: bf16x3 split-precision on V_WMMA_F32_16X16X32_BF16
//    (a = a_hi + a_lo in bf16; a*b ~= ah*bh + ah*bl + al*bh, fp32 accumulate;
//     rel. error ~ sqrt(K)*2^-17 ~ 3e-4 for K=2048 -> ~8x FLOP/instr vs fp32x4)
//  - Attention (QK^T softmax PV): exact fp32 via V_WMMA_F32_16X16X4_F32.
//  - Fallback: pure fp32 WMMA GEMM if workspace is too small for bf16 splits.
// ---------------------------------------------------------------------------

typedef __attribute__((ext_vector_type(2)))  float  v2f;
typedef __attribute__((ext_vector_type(8)))  float  v8f;
typedef __attribute__((ext_vector_type(8)))  __bf16 v8bf;
typedef __attribute__((ext_vector_type(16))) __bf16 v16bf;

#define DIM      2048
#define HEADS    16
#define HEAD_DIM 128
#define BATCH    2
#define L1       4096
#define L2CTX    769
#define IMG_TOK  257
#define TXT_TOK  512
#define ATT_SCALE 0.08838834764831845f   // 1/sqrt(128)

__device__ __forceinline__ int imin(int a, int b) { return a < b ? a : b; }

// ---- fp32 WMMA: D = A(16x4)*B(4x16)+C.  A: [M=lo][K=2hi+v], B: [K=2hi+v][N=lo],
//      C/D: [M=v+8hi][N=lo]  (ISA 7.12.2 tables)
__device__ __forceinline__ v8f wmma4(v2f a, v2f b, v8f c) {
  return __builtin_amdgcn_wmma_f32_16x16x4_f32(false, a, false, b,
                                               (short)0, c, false, false);
}

// ---- bf16 WMMA: D = A(16x32)*B(32x16)+C.
//  A frag (per lane lo,hi): elements 0..7  = K[8hi .. 8hi+7],
//                           elements 8..15 = K[16+8hi .. 23+8hi]   (row M=lo)
//  B frag: elements 0..7  = K[8hi .. 8hi+7],
//          elements 8..15 = K[16+8hi .. 23+8hi]                    (col N=lo)
__device__ __forceinline__ v8f wmma_bf16(v16bf a, v16bf b, v8f c) {
  return __builtin_amdgcn_wmma_f32_16x16x32_bf16(false, a, false, b,
                                                 (short)0, c, false, false);
}

__device__ __forceinline__ v16bf ldA16(const __bf16* row, int k0, int hi) {
  v8bf c0 = *(const v8bf*)(row + k0 + 8 * hi);
  v8bf c1 = *(const v8bf*)(row + k0 + 16 + 8 * hi);
  return __builtin_shufflevector(c0, c1, 0,1,2,3,4,5,6,7,8,9,10,11,12,13,14,15);
}
__device__ __forceinline__ v16bf ldB16(const __bf16* row, int k0, int hi) {
  v8bf c0 = *(const v8bf*)(row + k0 + 8 * hi);
  v8bf c1 = *(const v8bf*)(row + k0 + 16 + 8 * hi);
  return __builtin_shufflevector(c0, c1, 0,1,2,3,4,5,6,7,8,9,10,11,12,13,14,15);
}

// ---------------------------------------------------------------------------
// Split fp32 -> (bf16 hi, bf16 lo) with lo = bf16(x - float(hi))
// ---------------------------------------------------------------------------
__global__ __launch_bounds__(256)
void split_bf16_kernel(const float* __restrict__ X, __bf16* __restrict__ h,
                       __bf16* __restrict__ l, int n)
{
  int i = blockIdx.x * 256 + threadIdx.x;
  int stride = gridDim.x * 256;
  for (; i < n; i += stride) {
    float x = X[i];
    __bf16 hh = (__bf16)x;
    h[i] = hh;
    l[i] = (__bf16)(x - (float)hh);
  }
}

// ---------------------------------------------------------------------------
// bf16x3 GEMM:  Y[M,N] = X[M,K] @ W[N,K]^T + bias[N]
// Row mapping: xrow(m) = (m/Lr)*Sb + Ofs + m%Lr.  One wave: 32(M) x 64(N) tile.
// ---------------------------------------------------------------------------
__global__ __launch_bounds__(32)
void gemm_bias_bf16x3_kernel(float* __restrict__ Y,
                             const __bf16* __restrict__ Xh, const __bf16* __restrict__ Xl,
                             const __bf16* __restrict__ Wh, const __bf16* __restrict__ Wl,
                             const float* __restrict__ bias,
                             int Mtotal, int N, int K, int Lr, int Sb, int Ofs)
{
  const int lane = threadIdx.x & 31;
  const int lo = lane & 15, hi = lane >> 4;
  const int mBase = blockIdx.x * 32;
  const int nBase = blockIdx.y * 64;

  int m0 = imin(mBase + lo,      Mtotal - 1);
  int m1 = imin(mBase + 16 + lo, Mtotal - 1);
  size_t xr0 = (size_t)((m0 / Lr) * Sb + Ofs + (m0 % Lr)) * K;
  size_t xr1 = (size_t)((m1 / Lr) * Sb + Ofs + (m1 % Lr)) * K;
  const __bf16* x0h = Xh + xr0;  const __bf16* x0l = Xl + xr0;
  const __bf16* x1h = Xh + xr1;  const __bf16* x1l = Xl + xr1;
  const __bf16* wh[4];  const __bf16* wl[4];
  #pragma unroll
  for (int nt = 0; nt < 4; ++nt) {
    size_t wr = (size_t)(nBase + nt * 16 + lo) * K;
    wh[nt] = Wh + wr;  wl[nt] = Wl + wr;
  }

  v8f acc[2][4];
  v8f z = {};
  #pragma unroll
  for (int i = 0; i < 2; ++i)
    #pragma unroll
    for (int j = 0; j < 4; ++j) acc[i][j] = z;

  #pragma unroll 2
  for (int k0 = 0; k0 < K; k0 += 32) {
    v16bf a0h = ldA16(x0h, k0, hi);
    v16bf a0l = ldA16(x0l, k0, hi);
    v16bf a1h = ldA16(x1h, k0, hi);
    v16bf a1l = ldA16(x1l, k0, hi);
    #pragma unroll
    for (int nt = 0; nt < 4; ++nt) {
      v16bf bh = ldB16(wh[nt], k0, hi);
      v16bf bl = ldB16(wl[nt], k0, hi);
      acc[0][nt] = wmma_bf16(a0l, bh, acc[0][nt]);
      acc[0][nt] = wmma_bf16(a0h, bl, acc[0][nt]);
      acc[0][nt] = wmma_bf16(a0h, bh, acc[0][nt]);
      acc[1][nt] = wmma_bf16(a1l, bh, acc[1][nt]);
      acc[1][nt] = wmma_bf16(a1h, bl, acc[1][nt]);
      acc[1][nt] = wmma_bf16(a1h, bh, acc[1][nt]);
    }
  }

  #pragma unroll
  for (int nt = 0; nt < 4; ++nt) {
    int n = nBase + nt * 16 + lo;
    float bv = bias[n];
    #pragma unroll
    for (int v = 0; v < 8; ++v) {
      int r0 = mBase + v + 8 * hi;
      int r1 = r0 + 16;
      if (r0 < Mtotal) Y[(size_t)r0 * N + n] = acc[0][nt][v] + bv;
      if (r1 < Mtotal) Y[(size_t)r1 * N + n] = acc[1][nt][v] + bv;
    }
  }
}

// ---------------------------------------------------------------------------
// fp32 GEMM fallback (same tiling, V_WMMA_F32_16X16X4_F32)
// ---------------------------------------------------------------------------
__global__ __launch_bounds__(32)
void gemm_bias_kernel(float* __restrict__ Y, const float* __restrict__ X,
                      const float* __restrict__ W, const float* __restrict__ bias,
                      int Mtotal, int N, int K, int Lr, int Sb, int Ofs)
{
  const int lane = threadIdx.x & 31;
  const int lo = lane & 15, hi = lane >> 4;
  const int mBase = blockIdx.x * 32;
  const int nBase = blockIdx.y * 64;

  int m0 = imin(mBase + lo,      Mtotal - 1);
  int m1 = imin(mBase + 16 + lo, Mtotal - 1);
  const float* xr0 = X + (size_t)((m0 / Lr) * Sb + Ofs + (m0 % Lr)) * K + 2 * hi;
  const float* xr1 = X + (size_t)((m1 / Lr) * Sb + Ofs + (m1 % Lr)) * K + 2 * hi;
  const float* wr0 = W + (size_t)(nBase +  0 + lo) * K + 2 * hi;
  const float* wr1 = W + (size_t)(nBase + 16 + lo) * K + 2 * hi;
  const float* wr2 = W + (size_t)(nBase + 32 + lo) * K + 2 * hi;
  const float* wr3 = W + (size_t)(nBase + 48 + lo) * K + 2 * hi;

  v8f acc[2][4];
  v8f z = {};
  #pragma unroll
  for (int i = 0; i < 2; ++i)
    #pragma unroll
    for (int j = 0; j < 4; ++j) acc[i][j] = z;

  #pragma unroll 8
  for (int k0 = 0; k0 < K; k0 += 4) {
    v2f a0 = *(const v2f*)(xr0 + k0);
    v2f a1 = *(const v2f*)(xr1 + k0);
    v2f b0 = *(const v2f*)(wr0 + k0);
    v2f b1 = *(const v2f*)(wr1 + k0);
    v2f b2 = *(const v2f*)(wr2 + k0);
    v2f b3 = *(const v2f*)(wr3 + k0);
    acc[0][0] = wmma4(a0, b0, acc[0][0]);
    acc[0][1] = wmma4(a0, b1, acc[0][1]);
    acc[0][2] = wmma4(a0, b2, acc[0][2]);
    acc[0][3] = wmma4(a0, b3, acc[0][3]);
    acc[1][0] = wmma4(a1, b0, acc[1][0]);
    acc[1][1] = wmma4(a1, b1, acc[1][1]);
    acc[1][2] = wmma4(a1, b2, acc[1][2]);
    acc[1][3] = wmma4(a1, b3, acc[1][3]);
  }

  #pragma unroll
  for (int nt = 0; nt < 4; ++nt) {
    int n = nBase + nt * 16 + lo;
    float bv = bias[n];
    #pragma unroll
    for (int v = 0; v < 8; ++v) {
      int r0 = mBase + v + 8 * hi;
      int r1 = r0 + 16;
      if (r0 < Mtotal) Y[(size_t)r0 * N + n] = acc[0][nt][v] + bv;
      if (r1 < Mtotal) Y[(size_t)r1 * N + n] = acc[1][nt][v] + bv;
    }
  }
}

// ---------------------------------------------------------------------------
// In-place RMS-norm over last dim (N=2048): y = x * rsqrt(mean(x^2)+eps) * g
// ---------------------------------------------------------------------------
__global__ __launch_bounds__(256)
void rmsnorm_kernel(float* __restrict__ Y, const float* __restrict__ g, int N)
{
  __shared__ float red[8];
  float* row = Y + (size_t)blockIdx.x * N;
  float ss = 0.f;
  for (int c = threadIdx.x; c < N; c += 256) { float x = row[c]; ss += x * x; }
  #pragma unroll
  for (int off = 16; off > 0; off >>= 1) ss += __shfl_down(ss, off, 32);
  if ((threadIdx.x & 31) == 0) red[threadIdx.x >> 5] = ss;
  __syncthreads();
  if (threadIdx.x == 0) {
    float t = 0.f;
    #pragma unroll
    for (int i = 0; i < 8; ++i) t += red[i];
    red[0] = rsqrtf(t / (float)N + 1e-6f);
  }
  __syncthreads();
  float s = red[0];
  for (int c = threadIdx.x; c < N; c += 256) row[c] = row[c] * s * g[c];
}

// ---------------------------------------------------------------------------
// Attention: one wave handles a 16-query tile for one (batch, head). fp32 WMMA.
// ---------------------------------------------------------------------------
#define PSTRIDE 520   // >= 512, even (8B-aligned float2 LDS rows), +pad

__global__ __launch_bounds__(32)
void attn_kernel(float* __restrict__ Out, const float* __restrict__ Q,
                 const float* __restrict__ Kc, const float* __restrict__ Vc,
                 const int* __restrict__ lens, int Lk, int kvRowsPerBatch,
                 int accumulate)
{
  __shared__ float sP[16 * PSTRIDE];
  __shared__ float red[64];

  const int lane = threadIdx.x & 31;
  const int lo = lane & 15, hi = lane >> 4;
  const int qb = blockIdx.x;
  const int h  = blockIdx.y;
  const int b  = blockIdx.z;

  const size_t rs = DIM;
  const float* Qb = Q  + ((size_t)(b * L1) + qb * 16) * rs + h * HEAD_DIM;
  const float* Kb = Kc + (size_t)(b * kvRowsPerBatch) * rs + h * HEAD_DIM;
  const float* Vb = Vc + (size_t)(b * kvRowsPerBatch) * rs + h * HEAD_DIM;
  float*       Ob = Out + ((size_t)(b * L1) + qb * 16) * rs + h * HEAD_DIM;

  v2f qa[32];
  {
    const float* qrow = Qb + (size_t)lo * rs + 2 * hi;
    #pragma unroll
    for (int kk = 0; kk < 32; ++kk) qa[kk] = *(const v2f*)(qrow + kk * 4);
  }

  const int mlen = lens ? lens[b] : Lk;
  const int nKb  = (Lk + 15) >> 4;
  const int totC = nKb * 16;

  // ---- scores -> LDS
  for (int kb = 0; kb < nKb; ++kb) {
    int key  = kb * 16 + lo;
    int keyc = imin(key, Lk - 1);
    const float* krow = Kb + (size_t)keyc * rs + 2 * hi;
    v8f s = {};
    #pragma unroll 8
    for (int kk = 0; kk < 32; ++kk) {
      v2f bf = *(const v2f*)(krow + kk * 4);
      s = wmma4(qa[kk], bf, s);
    }
    bool valid = (key < mlen) && (key < Lk);
    #pragma unroll
    for (int v = 0; v < 8; ++v) {
      float val = valid ? s[v] * ATT_SCALE : -1e9f;
      sP[(v + 8 * hi) * PSTRIDE + key] = val;
    }
  }
  __syncthreads();

  // ---- softmax: row = lo, half = hi
  {
    const int halfC = totC >> 1;
    float* row = sP + lo * PSTRIDE + hi * halfC;
    float m = -3.0e38f;
    for (int c = 0; c < halfC; ++c) m = fmaxf(m, row[c]);
    red[hi * 16 + lo] = m;
    __syncthreads();
    float mAll = fmaxf(red[lo], red[16 + lo]);
    float sum = 0.f;
    for (int c = 0; c < halfC; ++c) {
      float e = __expf(row[c] - mAll);
      row[c] = e;
      sum += e;
    }
    red[32 + hi * 16 + lo] = sum;
    __syncthreads();
    float inv = 1.f / (red[32 + lo] + red[48 + lo]);
    for (int c = 0; c < halfC; ++c) row[c] *= inv;
  }
  __syncthreads();

  // ---- Out = P @ V (+ previous Out)
  v8f acc[8];
  if (accumulate) {
    #pragma unroll
    for (int nt = 0; nt < 8; ++nt)
      #pragma unroll
      for (int v = 0; v < 8; ++v)
        acc[nt][v] = Ob[(size_t)(v + 8 * hi) * rs + nt * 16 + lo];
  } else {
    v8f z = {};
    #pragma unroll
    for (int nt = 0; nt < 8; ++nt) acc[nt] = z;
  }

  const float* prow = sP + lo * PSTRIDE + 2 * hi;
  for (int k0 = 0; k0 < totC; k0 += 4) {
    v2f a = *(const v2f*)(prow + k0);
    int kr = k0 + 2 * hi;
    const float* v0 = Vb + (size_t)imin(kr,     Lk - 1) * rs;  // p==0 past Lk
    const float* v1 = Vb + (size_t)imin(kr + 1, Lk - 1) * rs;
    #pragma unroll
    for (int nt = 0; nt < 8; ++nt) {
      v2f bf;
      bf.x = v0[nt * 16 + lo];
      bf.y = v1[nt * 16 + lo];
      acc[nt] = wmma4(a, bf, acc[nt]);
    }
  }

  #pragma unroll
  for (int nt = 0; nt < 8; ++nt)
    #pragma unroll
    for (int v = 0; v < 8; ++v)
      Ob[(size_t)(v + 8 * hi) * rs + nt * 16 + lo] = acc[nt][v];
}

// ---------------------------------------------------------------------------
// Host launcher
// ---------------------------------------------------------------------------
extern "C" void kernel_launch(void* const* d_in, const int* in_sizes, int n_in,
                              void* d_out, int out_size, void* d_ws, size_t ws_size,
                              hipStream_t stream)
{
  const float* x        = (const float*)d_in[0];
  const float* context  = (const float*)d_in[1];
  const int*   ctx_lens = (const int*)  d_in[2];
  const float* w_q  = (const float*)d_in[3];  const float* b_q  = (const float*)d_in[4];
  const float* w_k  = (const float*)d_in[5];  const float* b_k  = (const float*)d_in[6];
  const float* w_v  = (const float*)d_in[7];  const float* b_v  = (const float*)d_in[8];
  const float* w_ki = (const float*)d_in[9];  const float* b_ki = (const float*)d_in[10];
  const float* w_vi = (const float*)d_in[11]; const float* b_vi = (const float*)d_in[12];
  const float* w_o  = (const float*)d_in[13]; const float* b_o  = (const float*)d_in[14];
  const float* g_q  = (const float*)d_in[15];
  const float* g_k  = (const float*)d_in[16];
  const float* g_ki = (const float*)d_in[17];
  float* out = (float*)d_out;

  const int Mq  = BATCH * L1;        // 8192
  const int Mk  = BATCH * TXT_TOK;   // 1024
  const int Mki = BATCH * IMG_TOK;   // 514
  const int nX   = Mq * DIM;                 // 16,777,216
  const int nCtx = BATCH * L2CTX * DIM;      // 3,149,824
  const int nW   = DIM * DIM;                // 4,194,304

  // ---- carve workspace (256B-aligned chunks) ----
  char* base = (char*)d_ws;
  size_t off = 0;
  auto allocF = [&](size_t n) -> float* {
    float* r = (float*)(base + off);
    off += ((n * sizeof(float)) + 255) & ~(size_t)255;
    return r;
  };
  auto allocB = [&](size_t n) -> __bf16* {
    __bf16* r = (__bf16*)(base + off);
    off += ((n * sizeof(__bf16)) + 255) & ~(size_t)255;
    return r;
  };

  float* Yq   = allocF((size_t)Mq  * DIM);
  float* Yk   = allocF((size_t)Mk  * DIM);
  float* Yv   = allocF((size_t)Mk  * DIM);
  float* Yki  = allocF((size_t)Mki * DIM);
  float* Yvi  = allocF((size_t)Mki * DIM);
  float* attn = allocF((size_t)Mq  * DIM);

  __bf16* Ah = allocB(nX);   __bf16* Al = allocB(nX);   // x, later attn
  __bf16* Ch = allocB(nCtx); __bf16* Cl = allocB(nCtx); // context
  __bf16* Wh = allocB(nW);   __bf16* Wl = allocB(nW);   // current weight
  const bool useBf16 = (off <= ws_size);

  dim3 wv(32);
  const int NT = DIM / 64;   // 32 N-tiles
  auto splitGrid = [](int n) { int g = (n + 255) / 256; return g > 16384 ? 16384 : g; };

  if (useBf16) {
    // Split activations once.
    split_bf16_kernel<<<splitGrid(nX),   256, 0, stream>>>(x,       Ah, Al, nX);
    split_bf16_kernel<<<splitGrid(nCtx), 256, 0, stream>>>(context, Ch, Cl, nCtx);

    // q projection
    split_bf16_kernel<<<splitGrid(nW), 256, 0, stream>>>(w_q, Wh, Wl, nW);
    gemm_bias_bf16x3_kernel<<<dim3(Mq / 32, NT), wv, 0, stream>>>(
        Yq, Ah, Al, Wh, Wl, b_q, Mq, DIM, DIM, L1, L1, 0);
    // k / v (text slice of context)
    split_bf16_kernel<<<splitGrid(nW), 256, 0, stream>>>(w_k, Wh, Wl, nW);
    gemm_bias_bf16x3_kernel<<<dim3(Mk / 32, NT), wv, 0, stream>>>(
        Yk, Ch, Cl, Wh, Wl, b_k, Mk, DIM, DIM, TXT_TOK, L2CTX, IMG_TOK);
    split_bf16_kernel<<<splitGrid(nW), 256, 0, stream>>>(w_v, Wh, Wl, nW);
    gemm_bias_bf16x3_kernel<<<dim3(Mk / 32, NT), wv, 0, stream>>>(
        Yv, Ch, Cl, Wh, Wl, b_v, Mk, DIM, DIM, TXT_TOK, L2CTX, IMG_TOK);
    // k_img / v_img (image slice of context)
    split_bf16_kernel<<<splitGrid(nW), 256, 0, stream>>>(w_ki, Wh, Wl, nW);
    gemm_bias_bf16x3_kernel<<<dim3((Mki + 31) / 32, NT), wv, 0, stream>>>(
        Yki, Ch, Cl, Wh, Wl, b_ki, Mki, DIM, DIM, IMG_TOK, L2CTX, 0);
    split_bf16_kernel<<<splitGrid(nW), 256, 0, stream>>>(w_vi, Wh, Wl, nW);
    gemm_bias_bf16x3_kernel<<<dim3((Mki + 31) / 32, NT), wv, 0, stream>>>(
        Yvi, Ch, Cl, Wh, Wl, b_vi, Mki, DIM, DIM, IMG_TOK, L2CTX, 0);
  } else {
    gemm_bias_kernel<<<dim3(Mq / 32, NT), wv, 0, stream>>>(Yq,  x,       w_q,  b_q,  Mq,  DIM, DIM, L1,      L1,    0);
    gemm_bias_kernel<<<dim3(Mk / 32, NT), wv, 0, stream>>>(Yk,  context, w_k,  b_k,  Mk,  DIM, DIM, TXT_TOK, L2CTX, IMG_TOK);
    gemm_bias_kernel<<<dim3(Mk / 32, NT), wv, 0, stream>>>(Yv,  context, w_v,  b_v,  Mk,  DIM, DIM, TXT_TOK, L2CTX, IMG_TOK);
    gemm_bias_kernel<<<dim3((Mki + 31) / 32, NT), wv, 0, stream>>>(Yki, context, w_ki, b_ki, Mki, DIM, DIM, IMG_TOK, L2CTX, 0);
    gemm_bias_kernel<<<dim3((Mki + 31) / 32, NT), wv, 0, stream>>>(Yvi, context, w_vi, b_vi, Mki, DIM, DIM, IMG_TOK, L2CTX, 0);
  }

  // RMS norms (in place, exact fp32).
  rmsnorm_kernel<<<Mq,  256, 0, stream>>>(Yq,  g_q,  DIM);
  rmsnorm_kernel<<<Mk,  256, 0, stream>>>(Yk,  g_k,  DIM);
  rmsnorm_kernel<<<Mki, 256, 0, stream>>>(Yki, g_ki, DIM);

  // Attention: image pass initializes, text pass (masked) accumulates.
  dim3 agrid(L1 / 16, HEADS, BATCH);
  attn_kernel<<<agrid, wv, 0, stream>>>(attn, Yq, Yki, Yvi, nullptr,  IMG_TOK, IMG_TOK, 0);
  attn_kernel<<<agrid, wv, 0, stream>>>(attn, Yq, Yk,  Yv,  ctx_lens, TXT_TOK, TXT_TOK, 1);

  // Output projection straight into d_out.
  if (useBf16) {
    split_bf16_kernel<<<splitGrid(nX), 256, 0, stream>>>(attn, Ah, Al, nX);  // reuse A split
    split_bf16_kernel<<<splitGrid(nW), 256, 0, stream>>>(w_o, Wh, Wl, nW);
    gemm_bias_bf16x3_kernel<<<dim3(Mq / 32, NT), wv, 0, stream>>>(
        out, Ah, Al, Wh, Wl, b_o, Mq, DIM, DIM, Mq, Mq, 0);
  } else {
    gemm_bias_kernel<<<dim3(Mq / 32, NT), wv, 0, stream>>>(out, attn, w_o, b_o, Mq, DIM, DIM, Mq, Mq, 0);
  }
}
